// DeeperGCN_LineGraph_17729624997969
// MI455X (gfx1250) — matrix operand order repaired and checked by reading, not compiled, not run on hardware
//
#include <hip/hip_runtime.h>

// ---------------------------------------------------------------------------
// DeeperGCN line-graph pipeline for MI455X (gfx1250, wave32, WMMA).
// GEMMs run on v_wmma_f32_16x16x32_bf16 (bf16 operands, f32 accumulate).
// ---------------------------------------------------------------------------

typedef __bf16 bf16_t;
typedef bf16_t v16bf __attribute__((ext_vector_type(16)));
typedef bf16_t v8bf  __attribute__((ext_vector_type(8)));
typedef float  v8f   __attribute__((ext_vector_type(8)));

#define HDIM 256
#define NLAYERS 4

// ---------------------------------------------------------------- encoder
__global__ void encode_kernel(const float* __restrict__ xg,
                              const float* __restrict__ Wenc,
                              const float* __restrict__ benc,
                              float* __restrict__ hnode, int Nn) {
  int i = blockIdx.x; if (i >= Nn) return;
  int c = threadIdx.x;
  __shared__ float xs[16];
  if (c < 16) xs[c] = xg[(size_t)i * 16 + c];
  __syncthreads();
  float acc = benc[c];
#pragma unroll
  for (int k = 0; k < 16; ++k) acc += xs[k] * Wenc[k * HDIM + c];
  hnode[(size_t)i * HDIM + c] = acc;
}

// --------------------------------------------- weight -> WMMA B-fragment
// frag layout: [nTile][kStep][lane][16] bf16, matching B(32x16) lane layout:
// lane = half*16 + r ; column n = nTile*16 + r ; k = kStep*32 + (j<8 ? half*8 + j : 16 + half*8 + j-8)
__global__ void pack_wfrag_kernel(const float* __restrict__ W,
                                  bf16_t* __restrict__ frag,
                                  int Kreal, int Kpad, int Ncols) {
  const int Ks = Kpad >> 5;
  const long total = (long)Ncols * Kpad;
  for (long idx = (long)blockIdx.x * 256 + threadIdx.x; idx < total;
       idx += (long)gridDim.x * 256) {
    int j    = (int)(idx & 15);
    int lane = (int)((idx >> 4) & 31);
    int ks   = (int)((idx >> 9) % Ks);
    int nt   = (int)(idx / (512L * Ks));
    int half = lane >> 4, r = lane & 15;
    int n = nt * 16 + r;
    int k = ks * 32 + ((j < 8) ? half * 8 + j : 16 + half * 8 + (j - 8));
    float w = (k < Kreal) ? W[(size_t)k * Ncols + n] : 0.f;
    frag[idx] = (bf16_t)w;
  }
}

// ------------------------------------------------ 4-wide basis projection
__global__ void basis_kernel(const float* __restrict__ X,
                             const float* __restrict__ Wg,
                             const float* __restrict__ bg,
                             float* __restrict__ out, long rows) {
  long total = rows * 4;
  for (long idx = (long)blockIdx.x * 256 + threadIdx.x; idx < total;
       idx += (long)gridDim.x * 256) {
    int j = (int)(idx & 3);
    long i = idx >> 2;
    float a = bg[j];
#pragma unroll
    for (int k = 0; k < 4; ++k) a += X[i * 4 + k] * Wg[k * 4 + j];
    out[idx] = a;
  }
}

// ---------------------------------- msg concat row -> padded bf16 (K=544)
__global__ void pack_msg_kernel(const float* __restrict__ hnode,
                                const float* __restrict__ eattr,
                                const float* __restrict__ xlg,
                                const int* __restrict__ srcg,
                                const int* __restrict__ dstg,
                                bf16_t* __restrict__ Apack, int Eg) {
  int e = blockIdx.x; if (e >= Eg) return;
  int s = srcg[e], d = dstg[e];
  bf16_t* row = Apack + (size_t)e * 544;
  for (int c = threadIdx.x; c < 544; c += blockDim.x) {
    float v;
    if (c < 256)      v = hnode[(size_t)s * HDIM + c];
    else if (c < 512) v = hnode[(size_t)d * HDIM + (c - 256)];
    else if (c < 528) v = eattr[(size_t)e * 16 + (c - 512)];
    else if (c < 532) v = xlg[(size_t)e * 4 + (c - 528)];
    else              v = 0.f;
    row[c] = (bf16_t)v;
  }
}

// ---------------------------------------------------------- WMMA GEMM
// C[M,N] = act(A_bf16[M,Kpad] @ Bfrag + bias) (+resid). Wave -> 16x32 tile.
__global__ __launch_bounds__(128)
void wmma_gemm_kernel(const bf16_t* __restrict__ A,
                      const bf16_t* __restrict__ Bfrag,
                      const float* __restrict__ bias,
                      const float* __restrict__ resid,
                      float* __restrict__ outF, bf16_t* __restrict__ outB,
                      int M, int Kpad, int Ncols, int doRelu) {
  const int lane = threadIdx.x & 31;
  const int wave = threadIdx.x >> 5;
  const int half = lane >> 4;
  const int r    = lane & 15;
  const int n0    = blockIdx.x * 32;
  const int mTile = (blockIdx.y * 4 + wave) * 16;
  if (mTile >= M) return;
  const int Ks = Kpad >> 5;
  const bf16_t* bp0 = Bfrag + ((size_t)(blockIdx.x * 2    ) * Ks) * 512;
  const bf16_t* bp1 = Bfrag + ((size_t)(blockIdx.x * 2 + 1) * Ks) * 512;
  const bf16_t* ap  = A + (size_t)(mTile + r) * Kpad + half * 8;

  v8f acc0 = {}; v8f acc1 = {};
  for (int ks = 0; ks < Ks; ++ks) {
    union { v16bf v; v8bf h[2]; } a, b0, b1;
    a.h[0] = *reinterpret_cast<const v8bf*>(ap);        // K: kb + half*8 .. +7
    a.h[1] = *reinterpret_cast<const v8bf*>(ap + 16);   // K: kb+16+half*8 .. +7
    ap += 32;
    b0.v = *reinterpret_cast<const v16bf*>(bp0 + ((size_t)ks * 32 + lane) * 16);
    b1.v = *reinterpret_cast<const v16bf*>(bp1 + ((size_t)ks * 32 + lane) * 16);
    acc0 = __builtin_amdgcn_wmma_f32_16x16x32_bf16(false, a.v, false, b0.v,
                                                   (short)0, acc0, false, false);
    acc1 = __builtin_amdgcn_wmma_f32_16x16x32_bf16(false, a.v, false, b1.v,
                                                   (short)0, acc1, false, false);
  }
  const float bia0 = bias ? bias[n0 + r]      : 0.f;
  const float bia1 = bias ? bias[n0 + 16 + r] : 0.f;
#pragma unroll
  for (int v = 0; v < 8; ++v) {
    int row = mTile + v + half * 8;           // C layout: lanes0-15 M=v, 16-31 M=v+8
    size_t o0 = (size_t)row * Ncols + n0 + r;
    size_t o1 = o0 + 16;
    float x0 = acc0[v] + bia0;
    float x1 = acc1[v] + bia1;
    if (doRelu) { x0 = fmaxf(x0, 0.f); x1 = fmaxf(x1, 0.f); }
    if (resid)  { x0 += resid[o0];     x1 += resid[o1]; }
    if (outF)   { outF[o0] = x0;       outF[o1] = x1; }
    if (outB)   { outB[o0] = (bf16_t)x0; outB[o1] = (bf16_t)x1; }
  }
}

// --------------------------------------------- GEN-Conv softmax aggregation
__global__ void gen_pass1_kernel(const float* __restrict__ x,
                                 const float* __restrict__ nbg,
                                 const float* __restrict__ ebg,
                                 const float* __restrict__ Wlnb,
                                 const float* __restrict__ blnb,
                                 const float* __restrict__ Wleb,
                                 const float* __restrict__ bleb,
                                 const int* __restrict__ src,
                                 const int* __restrict__ dst,
                                 float* __restrict__ m,
                                 unsigned int* __restrict__ mmax, int Elg) {
  int e = blockIdx.x; if (e >= Elg) return;
  int c = threadIdx.x;
  int s = src[e], d = dst[e];
  float nb = blnb[c], eb = bleb[c];
#pragma unroll
  for (int j = 0; j < 4; ++j) {
    nb += nbg[(size_t)s * 4 + j] * Wlnb[j * HDIM + c];
    eb += ebg[(size_t)e * 4 + j] * Wleb[j * HDIM + c];
  }
  float mv = fmaxf(x[(size_t)s * HDIM + c] + nb + eb, 0.f) + 1e-7f;
  m[(size_t)e * HDIM + c] = mv;
  // mv > 0 always -> uint bit pattern is order-preserving
  atomicMax(&mmax[(size_t)d * HDIM + c], __float_as_uint(mv));
}

__global__ void gen_pass2_kernel(const float* __restrict__ m,
                                 const float* __restrict__ mmax,
                                 const int* __restrict__ dst,
                                 float* __restrict__ ssum,
                                 float* __restrict__ tagg, int Elg) {
  int e = blockIdx.x; if (e >= Elg) return;
  int c = threadIdx.x;
  int d = dst[e];
  float mv = m[(size_t)e * HDIM + c];
  float em = __expf(mv - mmax[(size_t)d * HDIM + c]);
  atomicAdd(&ssum[(size_t)d * HDIM + c], em);
  atomicAdd(&tagg[(size_t)d * HDIM + c], em * mv);
}

__global__ void gen_pass3_kernel(const float* __restrict__ xin,
                                 const float* __restrict__ ssum,
                                 const float* __restrict__ tagg,
                                 bf16_t* __restrict__ zbf, int Eg) {
  int i = blockIdx.x; if (i >= Eg) return;
  int c = threadIdx.x;
  size_t o = (size_t)i * HDIM + c;
  float aggr = tagg[o] / (ssum[o] + 1e-16f);
  zbf[o] = (bf16_t)(xin[o] + aggr);
}

// ----------------------------------------------------------- BatchNorm
__global__ void bn_stats_kernel(const float* __restrict__ h,
                                float* __restrict__ bsum,
                                float* __restrict__ bsq, int M) {
  int c = threadIdx.x;
  float ls = 0.f, lq = 0.f;
  for (int row = blockIdx.x; row < M; row += gridDim.x) {
    float v = h[(size_t)row * HDIM + c];
    ls += v; lq += v * v;
  }
  atomicAdd(&bsum[c], ls);
  atomicAdd(&bsq[c], lq);
}

__global__ void bn_final_kernel(const float* __restrict__ bsum,
                                const float* __restrict__ bsq,
                                const float* __restrict__ gamma,
                                const float* __restrict__ beta,
                                float* __restrict__ scale,
                                float* __restrict__ shift, int M) {
  int c = threadIdx.x;
  float mu  = bsum[c] / (float)M;
  float var = bsq[c] / (float)M - mu * mu;
  float sc  = gamma[c] * rsqrtf(var + 1e-5f);
  scale[c] = sc;
  shift[c] = beta[c] - mu * sc;
}

__global__ void bn_relu_kernel(const float* __restrict__ h,
                               const float* __restrict__ scale,
                               const float* __restrict__ shift,
                               float* __restrict__ h2, int M) {
  int i = blockIdx.x; if (i >= M) return;
  int c = threadIdx.x;
  size_t o = (size_t)i * HDIM + c;
  h2[o] = fmaxf(scale[c] * h[o] + shift[c], 0.f);
}

// ---------------------------------------------- readout: scatter/pool/pred
__global__ void scatter_nodes_kernel(const float* __restrict__ h,
                                     const float* __restrict__ scale,
                                     const float* __restrict__ shift,
                                     const int* __restrict__ dstg,
                                     float* __restrict__ nodeEmb, int Eg) {
  int e = blockIdx.x; if (e >= Eg) return;
  int c = threadIdx.x;
  int d = dstg[e];
  atomicAdd(&nodeEmb[(size_t)d * HDIM + c],
            scale[c] * h[(size_t)e * HDIM + c] + shift[c]);
}

__global__ void pool_kernel(const float* __restrict__ nodeEmb,
                            const int* __restrict__ batch,
                            float* __restrict__ gsum,
                            float* __restrict__ cnt, int Nn) {
  int n = blockIdx.x; if (n >= Nn) return;
  int c = threadIdx.x;
  int b = batch[n];
  atomicAdd(&gsum[(size_t)b * HDIM + c], nodeEmb[(size_t)n * HDIM + c]);
  if (c == 0) atomicAdd(&cnt[b], 1.f);
}

__global__ void pred_kernel(const float* __restrict__ gsum,
                            const float* __restrict__ cnt,
                            const float* __restrict__ Wpred,
                            const float* __restrict__ bpred,
                            float* __restrict__ out, int nGraphs) {
  int b = blockIdx.x; if (b >= nGraphs) return;
  int c = threadIdx.x;
  __shared__ float red[HDIM];
  float inv = 1.f / fmaxf(cnt[b], 1.f);
  red[c] = gsum[(size_t)b * HDIM + c] * inv * Wpred[c];
  __syncthreads();
  for (int s2 = HDIM / 2; s2 > 0; s2 >>= 1) {
    if (c < s2) red[c] += red[c + s2];
    __syncthreads();
  }
  if (c == 0) out[b] = red[0] + bpred[0];
}

// ===========================================================================
extern "C" void kernel_launch(void* const* d_in, const int* in_sizes, int n_in,
                              void* d_out, int out_size, void* d_ws, size_t ws_size,
                              hipStream_t stream) {
  const float* xg    = (const float*)d_in[0];
  const float* eag   = (const float*)d_in[1];
  const float* xlg   = (const float*)d_in[2];
  const float* edb   = (const float*)d_in[3];
  const float* ealg  = (const float*)d_in[4];
  const float* Wenc  = (const float*)d_in[5];
  const float* benc  = (const float*)d_in[6];
  const float* Wmsg  = (const float*)d_in[7];
  const float* bmsg  = (const float*)d_in[8];
  const float* Wgnb  = (const float*)d_in[9];
  const float* bgnb  = (const float*)d_in[10];
  const float* Wgeb  = (const float*)d_in[11];
  const float* bgeb  = (const float*)d_in[12];
  const float* Wlnb  = (const float*)d_in[13];
  const float* blnb  = (const float*)d_in[14];
  const float* Wleb  = (const float*)d_in[15];
  const float* bleb  = (const float*)d_in[16];
  const float* W1    = (const float*)d_in[17];
  const float* b1    = (const float*)d_in[18];
  const float* W2    = (const float*)d_in[19];
  const float* b2    = (const float*)d_in[20];
  const float* bng   = (const float*)d_in[21];
  const float* bnb   = (const float*)d_in[22];
  const float* Wpred = (const float*)d_in[23];
  const float* bpred = (const float*)d_in[24];
  const int*   eig   = (const int*)d_in[25];
  const int*   eilg  = (const int*)d_in[26];
  const int*   batch = (const int*)d_in[27];
  (void)n_in; (void)ws_size;

  const int Nn      = in_sizes[0] / 16;   // 20000
  const int Eg      = in_sizes[1] / 16;   // 200000
  const int Elg     = in_sizes[4] / 4;    // 400000
  const int nGraphs = out_size;           // 128

  const int* srcg = eig;   const int* dstg = eig + Eg;
  const int* srcl = eilg;  const int* dstl = eilg + Elg;

  char* p = (char*)d_ws;
  auto alloc = [&](size_t bytes) -> char* {
    char* rp = p; p += (bytes + 255) & ~(size_t)255; return rp;
  };
  float*  hnode   = (float*)alloc((size_t)Nn * HDIM * 4);
  bf16_t* Apack   = (bf16_t*)alloc((size_t)Eg * 544 * 2);
  bf16_t* WmsgF   = (bf16_t*)alloc((size_t)544 * HDIM * 2);
  bf16_t* W1F     = (bf16_t*)alloc((size_t)NLAYERS * HDIM * 512 * 2);
  bf16_t* W2F     = (bf16_t*)alloc((size_t)NLAYERS * 512 * HDIM * 2);
  float*  nbg     = (float*)alloc((size_t)Eg * 4 * 4);
  float*  ebg     = (float*)alloc((size_t)Elg * 4 * 4);
  float*  h       = (float*)alloc((size_t)Eg * HDIM * 4);
  float*  h2      = (float*)alloc((size_t)Eg * HDIM * 4);
  float*  mbuf    = (float*)alloc((size_t)Elg * HDIM * 4);
  float*  mmax    = (float*)alloc((size_t)Eg * HDIM * 4);
  float*  ssum    = (float*)alloc((size_t)Eg * HDIM * 4);
  float*  tagg    = (float*)alloc((size_t)Eg * HDIM * 4);
  bf16_t* zbf     = (bf16_t*)alloc((size_t)Eg * HDIM * 2);
  bf16_t* hidbf   = (bf16_t*)alloc((size_t)Eg * 512 * 2);
  float*  bsum    = (float*)alloc(HDIM * 4);
  float*  bsq     = (float*)alloc(HDIM * 4);
  float*  scale   = (float*)alloc(HDIM * 4);
  float*  shift   = (float*)alloc(HDIM * 4);
  float*  nodeEmb = (float*)alloc((size_t)Nn * HDIM * 4);
  float*  gsum    = (float*)alloc((size_t)nGraphs * HDIM * 4);
  float*  cnt     = (float*)alloc((size_t)nGraphs * 4);

  // 1) node encoder
  encode_kernel<<<Nn, HDIM, 0, stream>>>(xg, Wenc, benc, hnode, Nn);

  // 2) weight fragments (bf16, WMMA B layout)
  pack_wfrag_kernel<<<(544 * HDIM + 255) / 256, 256, 0, stream>>>(Wmsg, WmsgF, 532, 544, HDIM);
  for (int l = 0; l < NLAYERS; ++l) {
    pack_wfrag_kernel<<<(HDIM * 512 + 255) / 256, 256, 0, stream>>>(
        W1 + (size_t)l * HDIM * 512, W1F + (size_t)l * HDIM * 512, HDIM, HDIM, 512);
    pack_wfrag_kernel<<<(512 * HDIM + 255) / 256, 256, 0, stream>>>(
        W2 + (size_t)l * 512 * HDIM, W2F + (size_t)l * 512 * HDIM, 512, 512, HDIM);
  }

  // 3) bottleneck basis projections
  basis_kernel<<<((long)Eg * 4 + 255) / 256, 256, 0, stream>>>(edb, Wgnb, bgnb, nbg, Eg);
  basis_kernel<<<((long)Elg * 4 + 255) / 256, 256, 0, stream>>>(ealg, Wgeb, bgeb, ebg, Elg);

  // 4) message concat + first GEMM:  h = concat @ W_msg + b_msg
  pack_msg_kernel<<<Eg, 256, 0, stream>>>(hnode, eag, xlg, srcg, dstg, Apack, Eg);
  {
    dim3 g(HDIM / 32, (Eg + 63) / 64);
    wmma_gemm_kernel<<<g, 128, 0, stream>>>(Apack, WmsgF, bmsg, nullptr,
                                            h, nullptr, Eg, 544, HDIM, 0);
  }

  // 5) DeeperGCN layers
  for (int l = 0; l < NLAYERS; ++l) {
    const float* xin = h;
    if (l > 0) {
      hipMemsetAsync(bsum, 0, HDIM * 4, stream);
      hipMemsetAsync(bsq,  0, HDIM * 4, stream);
      bn_stats_kernel<<<2048, HDIM, 0, stream>>>(h, bsum, bsq, Eg);
      bn_final_kernel<<<1, HDIM, 0, stream>>>(bsum, bsq, bng + (size_t)(l - 1) * HDIM,
                                              bnb + (size_t)(l - 1) * HDIM, scale, shift, Eg);
      bn_relu_kernel<<<Eg, HDIM, 0, stream>>>(h, scale, shift, h2, Eg);
      xin = h2;
    }
    hipMemsetAsync(mmax, 0, (size_t)Eg * HDIM * 4, stream);
    hipMemsetAsync(ssum, 0, (size_t)Eg * HDIM * 4, stream);
    hipMemsetAsync(tagg, 0, (size_t)Eg * HDIM * 4, stream);
    gen_pass1_kernel<<<Elg, HDIM, 0, stream>>>(xin, nbg, ebg,
        Wlnb + (size_t)l * 4 * HDIM, blnb + (size_t)l * HDIM,
        Wleb + (size_t)l * 4 * HDIM, bleb + (size_t)l * HDIM,
        srcl, dstl, mbuf, (unsigned int*)mmax, Elg);
    gen_pass2_kernel<<<Elg, HDIM, 0, stream>>>(mbuf, mmax, dstl, ssum, tagg, Elg);
    gen_pass3_kernel<<<Eg, HDIM, 0, stream>>>(xin, ssum, tagg, zbf, Eg);

    dim3 g1(512 / 32, (Eg + 63) / 64);
    wmma_gemm_kernel<<<g1, 128, 0, stream>>>(zbf, W1F + (size_t)l * HDIM * 512,
                                             b1 + (size_t)l * 512, nullptr,
                                             nullptr, hidbf, Eg, HDIM, 512, 1);
    dim3 g2(HDIM / 32, (Eg + 63) / 64);
    wmma_gemm_kernel<<<g2, 128, 0, stream>>>(hidbf, W2F + (size_t)l * 512 * HDIM,
                                             b2 + (size_t)l * HDIM,
                                             (l > 0) ? h : nullptr,
                                             h, nullptr, Eg, 512, HDIM, 0);
  }

  // 6) final BN (fused into node scatter) + pooling + prediction
  hipMemsetAsync(bsum, 0, HDIM * 4, stream);
  hipMemsetAsync(bsq,  0, HDIM * 4, stream);
  bn_stats_kernel<<<2048, HDIM, 0, stream>>>(h, bsum, bsq, Eg);
  bn_final_kernel<<<1, HDIM, 0, stream>>>(bsum, bsq, bng + (size_t)(NLAYERS - 1) * HDIM,
                                          bnb + (size_t)(NLAYERS - 1) * HDIM, scale, shift, Eg);
  hipMemsetAsync(nodeEmb, 0, (size_t)Nn * HDIM * 4, stream);
  scatter_nodes_kernel<<<Eg, HDIM, 0, stream>>>(h, scale, shift, dstg, nodeEmb, Eg);
  hipMemsetAsync(gsum, 0, (size_t)nGraphs * HDIM * 4, stream);
  hipMemsetAsync(cnt,  0, (size_t)nGraphs * 4, stream);
  pool_kernel<<<Nn, HDIM, 0, stream>>>(nodeEmb, batch, gsum, cnt, Nn);
  pred_kernel<<<nGraphs, HDIM, 0, stream>>>(gsum, cnt, Wpred, bpred, (float*)d_out, nGraphs);
}
